// CAM_87892210745945
// MI455X (gfx1250) — matrix operand order repaired and checked by reading, not compiled, not run on hardware
//
#include <hip/hip_runtime.h>
#include <hip/hip_bf16.h>

// Channel self-attention: E = X X^T, A = softmax(E), out = mu * (A X) + X
// X per batch: [512, 16384] f32 (32 MB -> L2 resident on MI455X's 192 MB L2).
// Matmuls run through v_wmma_f32_16x16x32_bf16 (bf16 in, f32 accum), with
// LDS double-buffering (pipeline unrolled x2, fixed buffer pointers) and the
// attention tile DMA'd by the Tensor Data Mover when available.

#define NB 8
#define CC 512
#define HH 16384
#define LSTR 40   // LDS row stride in bf16 elements (80 B, 16B-aligned rows)

typedef __bf16 bf16_t;
typedef bf16_t v16bf __attribute__((ext_vector_type(16)));
typedef float  v8f   __attribute__((ext_vector_type(8)));

struct Q32 { uint4 a, b; };  // 32 bytes == sizeof(v16bf)

#if defined(__has_builtin)
#if __has_builtin(__builtin_amdgcn_cvt_pk_bf16_f32)
#define HAVE_CVT_PK_BF16 1
#endif
#if __has_builtin(__builtin_amdgcn_tensor_load_to_lds) && \
    __has_builtin(__builtin_amdgcn_s_wait_tensorcnt)
#define HAVE_TDM 1
#endif
#endif

static __device__ __forceinline__ unsigned short f2bf(float f) {
    unsigned int u = __builtin_bit_cast(unsigned int, f);
    u += 0x7FFFu + ((u >> 16) & 1u);
    return (unsigned short)(u >> 16);
}

// pack bf16(lo) | bf16(hi)<<16 into one dword (v_cvt_pk_bf16_f32)
static __device__ __forceinline__ unsigned pk2(float lo, float hi) {
#ifdef HAVE_CVT_PK_BF16
    typedef __bf16 v2bf __attribute__((ext_vector_type(2)));
    v2bf v = __builtin_amdgcn_cvt_pk_bf16_f32(lo, hi);
    return __builtin_bit_cast(unsigned, v);
#else
    typedef __bf16 v2bf __attribute__((ext_vector_type(2)));
    v2bf v;
    v.x = (__bf16)lo;
    v.y = (__bf16)hi;
    return __builtin_bit_cast(unsigned, v);
#endif
}

static __device__ __forceinline__ uint2 pk4(float4 f) {
    uint2 r;
    r.x = pk2(f.x, f.y);
    r.y = pk2(f.z, f.w);
    return r;
}

// Build a 16xbf16 WMMA fragment from two 16-byte LDS runs
static __device__ __forceinline__ v16bf ld_frag(const unsigned short* p0,
                                                const unsigned short* p1) {
    Q32 q;
    q.a = *(const uint4*)p0;
    q.b = *(const uint4*)p1;
    return __builtin_bit_cast(v16bf, q);
}

// ---- CDNA5 async global->LDS copy (ASYNCcnt-tracked) ---------------------
static __device__ __forceinline__ void async_ld_b128(unsigned lds_off,
                                                     const void* gaddr) {
    asm volatile("global_load_async_to_lds_b128 %0, %1, off"
                 :: "v"(lds_off), "v"(gaddr) : "memory");
}
static __device__ __forceinline__ void wait_async0() {
    asm volatile("s_wait_asynccnt 0x0" ::: "memory");
}

// ---- CDNA5 Tensor Data Mover: 64-row x 64B tile of the attention matrix --
#ifdef HAVE_TDM
typedef unsigned int u32x4 __attribute__((ext_vector_type(4)));
typedef int          i32x8 __attribute__((ext_vector_type(8)));
typedef int          i32x4 __attribute__((ext_vector_type(4)));

static __device__ __forceinline__ void tdm_load_attn(const void* gsrc,
                                                     unsigned lds_off) {
    unsigned long long ga = (unsigned long long)gsrc;
    u32x4 g0;
    g0[0] = 1u;                                  // count=1, user mode
    g0[1] = lds_off;                             // lds_addr (bytes)
    g0[2] = (unsigned)ga;                        // global_addr[31:0]
    g0[3] = (unsigned)(ga >> 32) | (2u << 30);   // global_addr hi | type=2
    i32x8 g1;
    // data_size=3 (8B units), pad_enable, pad_interval=3 (64B), pad_amount=3 (4 dw)
    g1[0] = (3 << 16) | (1 << 20) | (3 << 22) | (3 << 25);
    g1[1] = 128 << 16;                           // tensor_dim0 = 128 units (1 KB row)
    g1[2] = 512 << 16;                           // tensor_dim1 = 512 rows
    g1[3] = 8 << 16;                             // tile_dim0 = 8 units (64 B)
    g1[4] = 64;                                  // tile_dim1 = 64 rows
    g1[5] = 128;                                 // tensor_dim0_stride = 128 units
    g1[6] = 0;
    g1[7] = 0;
    // clang-23 / therock form: 6 args (groups 2,3 + spare group, then cpol)
    __builtin_amdgcn_tensor_load_to_lds(g0, g1, (i32x4)0, (i32x4)0, (i32x8)0, 0);
}
#endif

// ---- shared MMA step: 1 A-frag x 8 B-frags = 8 WMMAs per wave ------------
static __device__ __forceinline__ void mma_tile(const unsigned short* bufA,
                                                const unsigned short* bufB,
                                                int wave, int l, int hi,
                                                v8f acc[8]) {
    const int arow = (wave << 4) + l;
    v16bf a = ld_frag(&bufA[arow * LSTR + (hi << 3)],
                      &bufA[arow * LSTR + 16 + (hi << 3)]);
    const unsigned short* bp = &bufB[l * LSTR + (hi << 3)];
    v16bf b = ld_frag(bp, bp + 16);
    #pragma unroll
    for (int j = 0; j < 7; ++j) {               // load b[j+1] ahead of wmma j
        const unsigned short* bn = &bufB[(((j + 1) << 4) + l) * LSTR + (hi << 3)];
        v16bf nb = ld_frag(bn, bn + 16);
        acc[j] = __builtin_amdgcn_wmma_f32_16x16x32_bf16(
            false, a, false, b, (short)0, acc[j], false, false);
        b = nb;
    }
    acc[7] = __builtin_amdgcn_wmma_f32_16x16x32_bf16(
        false, a, false, b, (short)0, acc[7], false, false);
}

// ---------------------------------------------------------------------------
// Kernel 1: E[n] = X[n] X[n]^T   (64(i) x 128(j) tile per 4-wave workgroup)
// ---------------------------------------------------------------------------
static __device__ __forceinline__ void stage_gram(const float* __restrict__ Xi,
                                                  const float* __restrict__ Xj,
                                                  unsigned short* dA,
                                                  unsigned short* dB, int t) {
    float4 va[4], vb[8];
    #pragma unroll
    for (int r = 0; r < 4; ++r) {
        int u = t + (r << 7);
        va[r] = *(const float4*)(Xi + (size_t)(u >> 3) * HH + ((u & 7) << 2));
    }
    #pragma unroll
    for (int r = 0; r < 8; ++r) {
        int u = t + (r << 7);
        vb[r] = *(const float4*)(Xj + (size_t)(u >> 3) * HH + ((u & 7) << 2));
    }
    // branchless speculative prefetch of the next tile
    __builtin_prefetch(Xi + (size_t)(t & 63) * HH + 32, 0, 1);
    __builtin_prefetch(Xj + (size_t)t * HH + 32, 0, 1);
    #pragma unroll
    for (int r = 0; r < 4; ++r) {
        int u = t + (r << 7);
        *(uint2*)&dA[(u >> 3) * LSTR + ((u & 7) << 2)] = pk4(va[r]);
    }
    #pragma unroll
    for (int r = 0; r < 8; ++r) {
        int u = t + (r << 7);
        *(uint2*)&dB[(u >> 3) * LSTR + ((u & 7) << 2)] = pk4(vb[r]);
    }
}

__global__ __launch_bounds__(128) void cam_gram(const float* __restrict__ x,
                                                float* __restrict__ E) {
    __shared__ __align__(16) unsigned short sA[2][64 * LSTR];
    __shared__ __align__(16) unsigned short sB[2][128 * LSTR];

    const int t  = threadIdx.x;
    const int j0 = blockIdx.x << 7;          // 128-wide j block
    const int i0 = blockIdx.y << 6;          // 64-wide i block
    const int n  = blockIdx.z;
    const float* Xn = x + (size_t)n * CC * HH;
    const float* Xi = Xn + (size_t)i0 * HH;
    const float* Xj = Xn + (size_t)j0 * HH;

    const int wave = t >> 5;
    const int lane = t & 31;
    const int l    = lane & 15;
    const int hi   = lane >> 4;

    v8f acc[8] = {};

    stage_gram(Xi, Xj, sA[0], sB[0], t);
    __syncthreads();

    for (int k0 = 0; k0 < HH; k0 += 64) {    // pipeline unrolled x2
        stage_gram(Xi + k0 + 32, Xj + k0 + 32, sA[1], sB[1], t);
        mma_tile(sA[0], sB[0], wave, l, hi, acc);
        __syncthreads();
        if (k0 + 64 < HH)
            stage_gram(Xi + k0 + 64, Xj + k0 + 64, sA[0], sB[0], t);
        mma_tile(sA[1], sB[1], wave, l, hi, acc);
        __syncthreads();
    }

    float* En = E + (size_t)n * CC * CC;
    #pragma unroll
    for (int j = 0; j < 8; ++j)
        #pragma unroll
        for (int r = 0; r < 8; ++r) {
            int gr = i0 + (wave << 4) + (hi << 3) + r;
            int gc = j0 + (j << 4) + l;
            En[(size_t)gr * CC + gc] = acc[j][r];
        }
}

// ---------------------------------------------------------------------------
// Kernel 2: row softmax over 512-wide rows; writes bf16 attention
// ---------------------------------------------------------------------------
__global__ __launch_bounds__(256) void cam_softmax(const float* __restrict__ E,
                                                   unsigned short* __restrict__ A) {
    const int row = blockIdx.x;                 // 0..NB*CC-1
    const float* e = E + (size_t)row * CC;
    const int t = threadIdx.x;
    const int wid = t >> 5, lane = t & 31;
    __shared__ float red[8];

    float v0 = e[t];
    float v1 = e[t + 256];

    float m = fmaxf(v0, v1);
    #pragma unroll
    for (int off = 16; off > 0; off >>= 1) m = fmaxf(m, __shfl_xor(m, off, 32));
    if (lane == 0) red[wid] = m;
    __syncthreads();
    m = red[lane & 7];
    #pragma unroll
    for (int off = 4; off > 0; off >>= 1) m = fmaxf(m, __shfl_xor(m, off, 32));
    __syncthreads();

    float e0 = __expf(v0 - m);
    float e1 = __expf(v1 - m);
    float s = e0 + e1;
    #pragma unroll
    for (int off = 16; off > 0; off >>= 1) s += __shfl_xor(s, off, 32);
    if (lane == 0) red[wid] = s;
    __syncthreads();
    s = red[lane & 7];
    #pragma unroll
    for (int off = 4; off > 0; off >>= 1) s += __shfl_xor(s, off, 32);

    float inv = 1.0f / s;
    unsigned short* a = A + (size_t)row * CC;
    a[t]       = f2bf(e0 * inv);
    a[t + 256] = f2bf(e1 * inv);
}

// ---------------------------------------------------------------------------
// Kernel 3: out = mu * (A X) + X   (64(C) x 128(H) tile per workgroup)
// ---------------------------------------------------------------------------
static __device__ __forceinline__ void stage_apply(
        const unsigned short* __restrict__ Ag,   // attention tile origin (bf16)
        const float* __restrict__ Xg,            // x tile origin (f32)
        unsigned short* dA, unsigned short* dB, int t, int wave) {
#ifdef HAVE_TDM
    // One TDM descriptor moves the whole 64x64B attention tile into LDS
    // (padded 16B per row to LSTR). TDM ignores EXEC; issue from wave 0 only.
    if (wave == 0)
        tdm_load_attn(Ag, (unsigned)(unsigned long long)dA);
#else
    #pragma unroll
    for (int r = 0; r < 2; ++r) {
        int u   = t + (r << 7);          // 0..255 (16B chunks)
        int row = u >> 2;                // 0..63
        int c8  = (u & 3) << 3;          // 0,8,16,24 (elements)
        async_ld_b128((unsigned)(unsigned long long)&dA[row * LSTR + c8],
                      Ag + (size_t)row * CC + c8);
    }
#endif
    // X tile 32(K) x 128(H): loads first (clause), then cvt + transposed store
    float f[8][4];
    #pragma unroll
    for (int r = 0; r < 8; ++r) {
        int u  = t + (r << 7);           // 0..1023
        int h  = u & 127;
        int k4 = (u >> 7) << 2;          // 0,4,...,28
        const float* p = Xg + (size_t)k4 * HH + h;
        f[r][0] = p[0];
        f[r][1] = p[HH];
        f[r][2] = p[2 * HH];
        f[r][3] = p[3 * HH];
    }
    #pragma unroll
    for (int r = 0; r < 8; ++r) {
        int u  = t + (r << 7);
        int h  = u & 127;
        int k4 = (u >> 7) << 2;
        uint2 d;
        d.x = pk2(f[r][0], f[r][1]);
        d.y = pk2(f[r][2], f[r][3]);
        *(uint2*)&dB[h * LSTR + k4] = d;
    }
}

static __device__ __forceinline__ void wait_stage_apply() {
#ifdef HAVE_TDM
    __builtin_amdgcn_s_wait_tensorcnt(0);
#else
    wait_async0();
#endif
}

__global__ __launch_bounds__(128) void cam_apply(const float* __restrict__ x,
                                                 const unsigned short* __restrict__ A,
                                                 const float* __restrict__ pmu,
                                                 float* __restrict__ out) {
    __shared__ __align__(16) unsigned short sA[2][64 * LSTR];
    __shared__ __align__(16) unsigned short sB[2][128 * LSTR];

    const int t  = threadIdx.x;
    const int h0 = blockIdx.x << 7;          // 128-wide h block
    const int c0 = blockIdx.y << 6;          // 64-wide c block
    const int n  = blockIdx.z;
    const float* Xn = x + (size_t)n * CC * HH;
    const unsigned short* An = A + (size_t)n * CC * CC + (size_t)c0 * CC;

    const int wave = t >> 5;
    const int lane = t & 31;
    const int l    = lane & 15;
    const int hi   = lane >> 4;

    v8f acc[8] = {};

    stage_apply(An, Xn + h0, sA[0], sB[0], t, wave);
    wait_stage_apply();
    __syncthreads();

    for (int k0 = 0; k0 < CC; k0 += 64) {    // pipeline unrolled x2
        stage_apply(An + (k0 + 32), Xn + (size_t)(k0 + 32) * HH + h0,
                    sA[1], sB[1], t, wave);
        mma_tile(sA[0], sB[0], wave, l, hi, acc);
        wait_stage_apply();
        __syncthreads();
        if (k0 + 64 < CC)
            stage_apply(An + (k0 + 64), Xn + (size_t)(k0 + 64) * HH + h0,
                        sA[0], sB[0], t, wave);
        mma_tile(sA[1], sB[1], wave, l, hi, acc);
        wait_stage_apply();
        __syncthreads();
    }

    const float mu = *pmu;
    float* On = out + (size_t)n * CC * HH;
    #pragma unroll
    for (int j = 0; j < 8; ++j)
        #pragma unroll
        for (int r = 0; r < 8; ++r) {
            int gc = c0 + (wave << 4) + (hi << 3) + r;
            int gh = h0 + (j << 4) + l;
            size_t idx = (size_t)gc * HH + gh;
            On[idx] = fmaf(mu, acc[j][r], Xn[idx]);
        }
}

// ---------------------------------------------------------------------------
extern "C" void kernel_launch(void* const* d_in, const int* in_sizes, int n_in,
                              void* d_out, int out_size, void* d_ws, size_t ws_size,
                              hipStream_t stream) {
    const float* x   = (const float*)d_in[0];
    const float* pmu = (const float*)d_in[1];
    float* out = (float*)d_out;

    // workspace: E f32 [8,512,512] (8 MB) then attention bf16 [8,512,512] (4 MB)
    float* E = (float*)d_ws;
    unsigned short* Abf =
        (unsigned short*)((char*)d_ws + (size_t)NB * CC * CC * sizeof(float));

    cam_gram   <<<dim3(CC / 128, CC / 64, NB), 128, 0, stream>>>(x, E);
    cam_softmax<<<dim3(NB * CC),               256, 0, stream>>>(E, Abf);
    cam_apply  <<<dim3(HH / 128, CC / 64, NB), 128, 0, stream>>>(x, Abf, pmu, out);
}
